// MemoryEngineLayer_40054865002730
// MI455X (gfx1250) — matrix operand order repaired and checked by reading, not compiled, not run on hardware
//
#include <hip/hip_runtime.h>
#include <math.h>

// ---------------- problem constants ----------------
constexpr int kH    = 1024;   // hidden dim
constexpr int kMEM  = 256;    // active memory slots
constexpr int kS    = 272;    // total slots
constexpr int kSP   = 288;    // slots padded to multiple of 32
constexpr int kB    = 8;
constexpr int kT    = 4096;
constexpr int kBT   = kB * kT;          // 32768 rows
constexpr int kRANK = 10;
constexpr int kTOPK = 8;
constexpr float kGAMMA = 0.92f, kBETA = 0.08f, kPTS = 0.4f;

// ---------------- WMMA types ----------------
typedef __attribute__((ext_vector_type(16))) __bf16 v16bf;
typedef __attribute__((ext_vector_type(2)))  __bf16 v2bf;
typedef __attribute__((ext_vector_type(8)))  float  v8f;

union Frag { v16bf v; unsigned int u[8]; };

// native converts -> compiler emits packed v_cvt bf16 path (no integer emulation)
__device__ __forceinline__ unsigned int pack_bf16(float lo, float hi) {
    union { v2bf v; unsigned int u; } c;
    c.v.x = (__bf16)lo;
    c.v.y = (__bf16)hi;
    return c.u;
}
__device__ __forceinline__ unsigned short f2bf(float f) {
    union { v2bf v; unsigned int u; } c;
    c.v.x = (__bf16)f; c.v.y = (__bf16)0.f;
    return (unsigned short)(c.u & 0xffffu);
}

// ============================================================
// Kernel 0: basis f32 -> bf16, both layouts (both zero-padded to kSP).
//   Bbf [kH][kSP]   -> B-matrix of GEMM2 (K = slots)
//   BTbf[kSP][kH]   -> B-matrix of GEMM1 (K = hidden), rows 272..287 zero
// ============================================================
__global__ void basis_convert_kernel(const float* __restrict__ basis,
                                     unsigned short* __restrict__ Bbf,
                                     unsigned short* __restrict__ BTbf) {
    int i = blockIdx.x * blockDim.x + threadIdx.x;
    if (i >= kH * kSP) return;
    int h = i / kSP, s = i - h * kSP;
    float v = (s < kS) ? basis[h * kS + s] : 0.0f;
    unsigned short bb = f2bf(v);
    Bbf[i] = bb;
    BTbf[(size_t)s * kH + h] = bb;
}

// ============================================================
// Kernel 1: M[bt, s] = sum_h X[bt,h] * basis[h,s]
// One wave per 16x32 output strip (A fragment reused for 2 WMMAs).
// K = 1024 in steps of 32, bf16 WMMA f32-acc.
// ============================================================
__global__ __launch_bounds__(128)
void gemm_proj_kernel(const float* __restrict__ X,
                      const unsigned short* __restrict__ BTbf,
                      float* __restrict__ Mout) {
    const int lane = threadIdx.x & 31;
    const int wave = threadIdx.x >> 5;
    const int CP = kSP / 32;                        // 9 column pairs
    int tile = blockIdx.x * 4 + wave;
    int row_tile = tile / CP;
    int cp       = tile - row_tile * CP;
    if (row_tile >= kBT / 16) return;

    const int r  = lane & 15;
    const int hg = lane >> 4;
    const float*          Arow  = X    + (size_t)(row_tile * 16 + r) * kH;
    const unsigned short* Brow0 = BTbf + (size_t)(cp * 32 + r) * kH;
    const unsigned short* Brow1 = Brow0 + (size_t)16 * kH;

    v8f acc0 = {}, acc1 = {};
    for (int k0 = 0; k0 < kH; k0 += 32) {
        Frag a, b0, b1;
#pragma unroll
        for (int v = 0; v < 8; ++v) {
            // A k order used here (contiguous, same as B): lanes 0-15 K=0..15 ; lanes 16-31 K=16..31
            int ka = 2 * v + 16 * hg;
            float2 f = *(const float2*)(Arow + k0 + ka);
            a.u[v] = pack_bf16(f.x, f.y);
            // B k order used here (contiguous, same as A): lanes 0-15 K=0..15 ; lanes 16-31 K=16..31
            int kb = 2 * v + 16 * hg;
            b0.u[v] = *(const unsigned int*)(Brow0 + k0 + kb);
            b1.u[v] = *(const unsigned int*)(Brow1 + k0 + kb);
        }
        acc0 = __builtin_amdgcn_wmma_f32_16x16x32_bf16(
                   false, a.v, false, b0.v, (short)0, acc0, false, false);
        acc1 = __builtin_amdgcn_wmma_f32_16x16x32_bf16(
                   false, a.v, false, b1.v, (short)0, acc1, false, false);
    }
    // C layout: lanes 0-15 -> M = i ; lanes 16-31 -> M = i + 8 ; N = lane&15
    int col = cp * 32 + r;
#pragma unroll
    for (int i = 0; i < 8; ++i) {
        int row = row_tile * 16 + i + 8 * hg;
        Mout[(size_t)row * kSP + col]      = acc0[i];
        Mout[(size_t)row * kSP + col + 16] = acc1[i];
    }
}

// ============================================================
// Kernel 2: sequential scan over T. One workgroup per batch element,
// thread-per-slot, tape in registers, low-rank operators expanded as
// F (scale . (F^T v)) + d.v  -> 40 cross-block dot products per step.
// ============================================================
__global__ __launch_bounds__(kSP)
void scan_kernel(const float* __restrict__ Mbuf,
                 float* __restrict__ Rbuf,
                 const float* __restrict__ tape_re, const float* __restrict__ tape_im,
                 const float* __restrict__ eta_raw,
                 const float* __restrict__ torque_rotation,
                 const float* __restrict__ w_r, const float* __restrict__ breadth_gate,
                 const float* __restrict__ eps_factor, const float* __restrict__ eps_scale,
                 const float* __restrict__ eps_diag,
                 const float* __restrict__ pred_factor, const float* __restrict__ pred_scale,
                 const float* __restrict__ pred_diag) {
    const int tid  = threadIdx.x;
    const int lane = tid & 31;
    const int wave = tid >> 5;                       // 9 waves
    const int b    = blockIdx.x;
    const bool valid = tid < kS;
    const bool act   = tid < kMEM;

    __shared__ float smag[kSP];
    __shared__ float redp[9 * 40];
    __shared__ float red1[9];
    __shared__ float bcast[44];
    __shared__ float scl[2 * kRANK];

    float Fe[kRANK], Fw[kRANK];
    float c0 = 1.f, s0 = 0.f, wr = 0.f, bgv = 0.f, deps = 0.f, dwp = 0.f;
    float tre = 0.f, tim = 0.f;
    if (valid) {
        float rv = torque_rotation[tid];
        c0 = cosf(rv); s0 = sinf(rv);
        wr  = 1.f / (1.f + expf(-w_r[tid]));
        bgv = 1.f / (1.f + expf(-breadth_gate[tid]));
        deps = eps_diag[tid]; dwp = pred_diag[tid];
#pragma unroll
        for (int r = 0; r < kRANK; ++r) {
            Fe[r] = eps_factor[tid * kRANK + r];
            Fw[r] = pred_factor[tid * kRANK + r];
        }
    } else {
#pragma unroll
        for (int r = 0; r < kRANK; ++r) { Fe[r] = 0.f; Fw[r] = 0.f; }
    }
    if (act) { tre = tape_re[tid]; tim = tape_im[tid]; }
    if (tid < kRANK)               scl[tid] = eps_scale[tid];
    else if (tid < 2 * kRANK)      scl[tid] = pred_scale[tid - kRANK];
    const float eta = logf(1.f + expf(eta_raw[0]));

    // initial renorm of tape0 (aux slots already zero)
    float q = tre * tre + tim * tim;
#pragma unroll
    for (int off = 16; off >= 1; off >>= 1) q += __shfl_xor(q, off);
    if (lane == 0) red1[wave] = q;
    __syncthreads();
    if (tid == 0) {
        float s = 0.f;
        for (int w = 0; w < 9; ++w) s += red1[w];
        bcast[40] = 1.f / sqrtf(fmaxf(s, 1e-16f));
    }
    __syncthreads();
    { float rn = bcast[40]; tre *= rn; tim *= rn; }
    float pvre = tre, pvim = tim;                    // prev = tape0

    size_t idx = (size_t)b * kT * kSP + tid;
    for (int t = 0; t < kT; ++t, idx += kSP) {
        float m = valid ? Mbuf[idx] : 0.f;
        if (t + 1 < kT) __builtin_prefetch(Mbuf + idx + kSP, 0, 1);
        float mag = act ? fabsf(m) : 0.f;
        smag[tid] = mag;
        __syncthreads();                             // B1: smag ready

        // top-8 threshold computed entirely inside wave 0 (9 slots per lane)
        if (wave == 0) {
            float a[9];
#pragma unroll
            for (int w = 0; w < 9; ++w) a[w] = smag[lane + 32 * w];
            float mx = 0.f;
            for (int it = 0; it < kTOPK; ++it) {
                mx = a[0];
#pragma unroll
                for (int w = 1; w < 9; ++w) mx = fmaxf(mx, a[w]);
#pragma unroll
                for (int off = 16; off >= 1; off >>= 1) mx = fmaxf(mx, __shfl_xor(mx, off));
#pragma unroll
                for (int w = 0; w < 9; ++w) if (a[w] == mx) a[w] = -1.f;
            }
            if (lane == 0) bcast[41] = mx;           // 8th largest
        }
        __syncthreads();                             // B2: kth ready
        const float kth = bcast[41];
        float inj = (act && mag >= kth) ? m : 0.f;

        // rotated = tape * exp(i*rot)
        float rre = tre * c0 - tim * s0;
        float rim = tre * s0 + tim * c0;

        // 40 projections: {eps,pred} x {re,im} x rank
#pragma unroll
        for (int r = 0; r < kRANK; ++r) {
            float p0 = Fe[r] * rre, p1 = Fe[r] * rim;
            float p2 = Fw[r] * rre, p3 = Fw[r] * rim;
#pragma unroll
            for (int off = 16; off >= 1; off >>= 1) {
                p0 += __shfl_xor(p0, off); p1 += __shfl_xor(p1, off);
                p2 += __shfl_xor(p2, off); p3 += __shfl_xor(p3, off);
            }
            if (lane == 0) {
                float* d = &redp[wave * 40 + r * 4];
                d[0] = p0; d[1] = p1; d[2] = p2; d[3] = p3;
            }
        }
        __syncthreads();                             // B3: wave partials ready
        if (tid < 40) {
            float s = 0.f;
            for (int w = 0; w < 9; ++w) s += redp[w * 40 + tid];
            int r = tid >> 2, comp = tid & 3;
            float sc = (comp < 2) ? scl[r] : scl[kRANK + r];
            bcast[tid] = sc * s;
        }
        __syncthreads();                             // B4: scaled projections ready

        // expand: drive = F_eps (scaled proj) + d_eps.v ; pred similarly
        float drre = deps * rre, drim = deps * rim;
        float pqre = dwp * rre,  pqim = dwp * rim;
#pragma unroll
        for (int r = 0; r < kRANK; ++r) {
            drre += Fe[r] * bcast[4 * r + 0];
            drim += Fe[r] * bcast[4 * r + 1];
            pqre += Fw[r] * bcast[4 * r + 2];
            pqim += Fw[r] * bcast[4 * r + 3];
        }
        // new = g*rot + eta*drive + beta*inj + PTS*i*pred + wr*prev
        float nre = kGAMMA * rre + eta * drre + kBETA * inj - kPTS * pqim + wr * pvre;
        float nim = kGAMMA * rim + eta * drim              + kPTS * pqre + wr * pvim;
        if (!act) { nre = 0.f; nim = 0.f; }          // mask before renorm

        float qq = nre * nre + nim * nim;
#pragma unroll
        for (int off = 16; off >= 1; off >>= 1) qq += __shfl_xor(qq, off);
        if (lane == 0) red1[wave] = qq;
        __syncthreads();                             // B5
        if (tid == 0) {
            float s = 0.f;
            for (int w = 0; w < 9; ++w) s += red1[w];
            bcast[42] = 1.f / sqrtf(fmaxf(s, 1e-16f));
        }
        __syncthreads();                             // B6: norm ready
        float rn = bcast[42];
        nre *= rn; nim *= rn;

        pvre = tre; pvim = tim;                      // prev <- old tape
        tre = nre; tim = nim;                        // tape <- new

        Rbuf[idx] = valid ? bgv * nre : 0.f;         // pad columns -> 0 for GEMM2
    }
}

// ============================================================
// Kernel 3: Y[bt,h] = X[bt,h] + alpha * sum_s R[bt,s] * basis[h,s]
// One wave per 16x32 output strip, K = 288 in steps of 32.
// ============================================================
__global__ __launch_bounds__(128)
void gemm_out_kernel(const float* __restrict__ R,
                     const unsigned short* __restrict__ Bbf,
                     const float* __restrict__ X,
                     const float* __restrict__ alpha_p,
                     float* __restrict__ Y) {
    const int lane = threadIdx.x & 31;
    const int wave = threadIdx.x >> 5;
    const int CP = kH / 32;                          // 32 column pairs
    int tile = blockIdx.x * 4 + wave;
    int row_tile = tile / CP;
    int cp       = tile - row_tile * CP;
    if (row_tile >= kBT / 16) return;

    const int r  = lane & 15;
    const int hg = lane >> 4;
    const float*          Arow  = R   + (size_t)(row_tile * 16 + r) * kSP;
    const unsigned short* Brow0 = Bbf + (size_t)(cp * 32 + r) * kSP;
    const unsigned short* Brow1 = Brow0 + (size_t)16 * kSP;

    v8f acc0 = {}, acc1 = {};
    for (int k0 = 0; k0 < kSP; k0 += 32) {
        Frag a, b0, b1;
#pragma unroll
        for (int v = 0; v < 8; ++v) {
            int ka = 2 * v + 16 * hg;
            float2 f = *(const float2*)(Arow + k0 + ka);
            a.u[v] = pack_bf16(f.x, f.y);
            int kb = 2 * v + 16 * hg;
            b0.u[v] = *(const unsigned int*)(Brow0 + k0 + kb);
            b1.u[v] = *(const unsigned int*)(Brow1 + k0 + kb);
        }
        acc0 = __builtin_amdgcn_wmma_f32_16x16x32_bf16(
                   false, a.v, false, b0.v, (short)0, acc0, false, false);
        acc1 = __builtin_amdgcn_wmma_f32_16x16x32_bf16(
                   false, a.v, false, b1.v, (short)0, acc1, false, false);
    }
    const float alpha = alpha_p[0];
    int col = cp * 32 + r;
#pragma unroll
    for (int i = 0; i < 8; ++i) {
        int row = row_tile * 16 + i + 8 * hg;
        size_t o = (size_t)row * kH + col;
        Y[o]      = X[o]      + alpha * acc0[i];
        Y[o + 16] = X[o + 16] + alpha * acc1[i];
    }
}

// ============================================================
// host-side launcher
// ============================================================
extern "C" void kernel_launch(void* const* d_in, const int* in_sizes, int n_in,
                              void* d_out, int out_size, void* d_ws, size_t ws_size,
                              hipStream_t stream) {
    (void)in_sizes; (void)n_in; (void)out_size; (void)ws_size;
    const float* x     = (const float*)d_in[0];
    const float* basis = (const float*)d_in[1];
    const float* t_re  = (const float*)d_in[2];
    const float* t_im  = (const float*)d_in[3];
    const float* eta   = (const float*)d_in[4];
    const float* alpha = (const float*)d_in[5];
    const float* rot   = (const float*)d_in[6];
    const float* w_r   = (const float*)d_in[7];
    const float* bg    = (const float*)d_in[8];
    const float* ef    = (const float*)d_in[9];
    const float* es    = (const float*)d_in[10];
    const float* ed    = (const float*)d_in[11];
    const float* pf    = (const float*)d_in[12];
    const float* ps    = (const float*)d_in[13];
    const float* pd    = (const float*)d_in[14];
    float* out = (float*)d_out;

    // workspace carve-up (all offsets 256B aligned)
    char* ws = (char*)d_ws;
    unsigned short* BTbf = (unsigned short*)(ws);                       // 288*1024*2  = 589824
    unsigned short* Bbf  = (unsigned short*)(ws + 589824);              // 1024*288*2  = 589824
    float*          Mbuf = (float*)(ws + 589824 + 589824);              // 32768*288*4 = 37748736
    float*          Rbuf = (float*)(ws + 589824 + 589824 + 37748736);   // 32768*288*4

    // 0) basis -> bf16 (both layouts)
    basis_convert_kernel<<<(kH * kSP + 255) / 256, 256, 0, stream>>>(basis, Bbf, BTbf);

    // 1) M = X @ basis  (2048 row tiles x 9 col pairs, 4 waves/block)
    gemm_proj_kernel<<<(kBT / 16) * (kSP / 32) / 4, 128, 0, stream>>>(x, BTbf, Mbuf);

    // 2) sequential recurrence, one workgroup per batch element
    scan_kernel<<<kB, kSP, 0, stream>>>(Mbuf, Rbuf, t_re, t_im, eta, rot, w_r, bg,
                                        ef, es, ed, pf, ps, pd);

    // 3) Y = X + alpha * R @ basis^T  (2048 x 32 col pairs, 4 waves/block)
    gemm_out_kernel<<<(kBT / 16) * (kH / 32) / 4, 128, 0, stream>>>(Rbuf, Bbf, x, alpha, out);
}
